// ARDRBFKernel_12661563589010
// MI455X (gfx1250) — compile-verified, matrix operand order checked by reading, not run on hardware
//
#include <hip/hip_runtime.h>
#include <hip/hip_bf16.h>

// ARD-RBF for MI455X (gfx1250), wave32 + V_WMMA_F32_16X16X32_F16.
//   out[n,m] = exp(log_os) * exp(-0.5 * ||(x[n]-y[m])/l||^2)
// GEMM rewrite: d2 = x2[n] + y2[m] - 2*(xs.ys^T), clamp >= 0.
//
// Cross term on the fast 16-bit matrix pipe with error-compensated
// hi/lo f16 split: xs = xh + xl (xl = f32 residual), keep
//   xh*yh + xh*yl + xl*yh   (drop xl*yl ~ 2^-22),
// accumulated in f32 by the WMMA. 3x MACs but 8x K per instruction vs
// the f32 WMMA -> matrix time ~5us, under the 11us HBM store floor
// (256 MB output @ 23.3 TB/s). Output bypasses L2 (256MB > 192MB).

typedef __attribute__((ext_vector_type(16))) _Float16 v16h;
typedef __attribute__((ext_vector_type(8)))  _Float16 v8h;
typedef __attribute__((ext_vector_type(8)))  float    v8f;

#define D_DIM 64
#define TILE  128          // 128x128 output tile per 8-wave workgroup
#define LDSH  72           // halfs per LDS row: 144 B, 16B-aligned, pad vs bank conflicts

// A/B fragment for 16x16x32 f16: two contiguous 16B runs of one LDS row.
// Per lane (row = M or N = lane&15, half = lane>>4):
//   elems 0..7  = K: kb + 8*half + {0..7}
//   elems 8..15 = K: kb + 16 + 8*half + {0..7}
static __device__ __forceinline__ v16h load_frag(const _Float16* p) {
    v8h lo = *(const v8h*)(p);
    v8h hi = *(const v8h*)(p + 16);
    return __builtin_shufflevector(lo, hi, 0,1,2,3,4,5,6,7,8,9,10,11,12,13,14,15);
}

// ---------------------------------------------------------------------------
// prep: v = x*exp(-log_l); split v -> (f16 hi, f16 lo); row norms in f32.
// One 64-thread block per row of x or y.
// ---------------------------------------------------------------------------
__global__ __launch_bounds__(64)
void ard_prep_kernel(const float* __restrict__ x, const float* __restrict__ y,
                     const float* __restrict__ log_l, int N,
                     _Float16* __restrict__ xs_h, _Float16* __restrict__ xs_l,
                     _Float16* __restrict__ ys_h, _Float16* __restrict__ ys_l,
                     float* __restrict__ x2, float* __restrict__ y2) {
    __shared__ float s[D_DIM];
    const int d   = threadIdx.x;          // 0..63
    const int blk = blockIdx.x;
    const float invl = __expf(-log_l[d]);

    const float* src; _Float16* dh; _Float16* dl; float* nrm; int row;
    if (blk < N) { src = x; dh = xs_h; dl = xs_l; nrm = x2; row = blk; }
    else         { src = y; dh = ys_h; dl = ys_l; nrm = y2; row = blk - N; }

    const float v = src[(size_t)row * D_DIM + d] * invl;
    const _Float16 h = (_Float16)v;
    const _Float16 l = (_Float16)(v - (float)h);
    dh[(size_t)row * D_DIM + d] = h;
    dl[(size_t)row * D_DIM + d] = l;
    s[d] = v * v;
    __syncthreads();
    for (int off = 32; off > 0; off >>= 1) {
        if (d < off) s[d] += s[d + off];
        __syncthreads();
    }
    if (d == 0) nrm[row] = s[0];
}

// ---------------------------------------------------------------------------
// main: 128x128 tile, 8 waves, each wave -> 2x4 grid of 16x16 WMMA tiles,
// 3-term compensated f16 WMMA, fused exp epilogue, NT stores.
// ---------------------------------------------------------------------------
__global__ __launch_bounds__(256)
void ard_rbf_wmma_kernel(const _Float16* __restrict__ xs_h, const _Float16* __restrict__ xs_l,
                         const _Float16* __restrict__ ys_h, const _Float16* __restrict__ ys_l,
                         const float* __restrict__ x2, const float* __restrict__ y2,
                         const float* __restrict__ log_os,
                         float* __restrict__ out, int M) {
    __shared__ _Float16 lxh[TILE * LDSH];   // 18432 B each, 73728 B total
    __shared__ _Float16 lxl[TILE * LDSH];
    __shared__ _Float16 lyh[TILE * LDSH];
    __shared__ _Float16 lyl[TILE * LDSH];

    const int tid     = threadIdx.x;
    const int rowBase = blockIdx.y * TILE;
    const int colBase = blockIdx.x * TILE;

    // ---- stage 4 tiles (128 rows x 64 halfs) into LDS, 16B transactions ----
    {
        const int rr = tid >> 3;          // 0..31
        const int cc = (tid & 7) * 8;     // half offset 0..56 (16B chunks)
        #pragma unroll
        for (int p = 0; p < TILE; p += 32) {
            const int r = rr + p;
            const size_t gx = (size_t)(rowBase + r) * D_DIM + cc;
            const size_t gy = (size_t)(colBase + r) * D_DIM + cc;
            *(uint4*)(&lxh[r * LDSH + cc]) = *(const uint4*)(xs_h + gx);
            *(uint4*)(&lxl[r * LDSH + cc]) = *(const uint4*)(xs_l + gx);
            *(uint4*)(&lyh[r * LDSH + cc]) = *(const uint4*)(ys_h + gy);
            *(uint4*)(&lyl[r * LDSH + cc]) = *(const uint4*)(ys_l + gy);
        }
    }
    __syncthreads();

    const int wave = tid >> 5;            // 0..7
    const int lane = tid & 31;
    const int half = lane >> 4;           // K-phase selector
    const int l16  = lane & 15;           // M (A) / N (B) index
    const int wrow = (wave & 3) * 32;     // 0,32,64,96
    const int wcol = (wave >> 2) * 64;    // 0,64

    v8f acc[2][4] = {};                   // 2x4 tiles of 16x16 f32

    // ---- K loop: 2 blocks of K=32; 3 compensated terms per tile ----
    #pragma unroll
    for (int kb = 0; kb < D_DIM; kb += 32) {
        const int koff = kb + 8 * half;
        v16h axh[2], axl[2], byh[4], byl[4];
        #pragma unroll
        for (int ti = 0; ti < 2; ++ti) {
            const int base = (wrow + ti * 16 + l16) * LDSH + koff;
            axh[ti] = load_frag(&lxh[base]);
            axl[ti] = load_frag(&lxl[base]);
        }
        #pragma unroll
        for (int tj = 0; tj < 4; ++tj) {
            const int base = (wcol + tj * 16 + l16) * LDSH + koff;
            byh[tj] = load_frag(&lyh[base]);
            byl[tj] = load_frag(&lyl[base]);
        }
        #pragma unroll
        for (int ti = 0; ti < 2; ++ti)
            #pragma unroll
            for (int tj = 0; tj < 4; ++tj) {
                acc[ti][tj] = __builtin_amdgcn_wmma_f32_16x16x32_f16(
                    false, axh[ti], false, byh[tj], (short)0, acc[ti][tj], false, false);
                acc[ti][tj] = __builtin_amdgcn_wmma_f32_16x16x32_f16(
                    false, axh[ti], false, byl[tj], (short)0, acc[ti][tj], false, false);
                acc[ti][tj] = __builtin_amdgcn_wmma_f32_16x16x32_f16(
                    false, axl[ti], false, byh[tj], (short)0, acc[ti][tj], false, false);
            }
    }

    const float oscale = __expf(log_os[0]);

    // ---- fused epilogue: d2 = x2 + y2 - 2c; out = os * exp(-0.5*d2) ----
    #pragma unroll
    for (int ti = 0; ti < 2; ++ti) {
        float xn[8];
        #pragma unroll
        for (int r = 0; r < 8; ++r)
            xn[r] = x2[rowBase + wrow + ti * 16 + r + half * 8];
        #pragma unroll
        for (int tj = 0; tj < 4; ++tj) {
            const int col = colBase + wcol + tj * 16 + l16;
            const float yn = y2[col];
            #pragma unroll
            for (int r = 0; r < 8; ++r) {
                const int row = rowBase + wrow + ti * 16 + r + half * 8;
                float d2 = fmaf(-2.0f, acc[ti][tj][r], xn[r] + yn);
                d2 = fmaxf(d2, 0.0f);
                const float val = oscale * __expf(-0.5f * d2);
                __builtin_nontemporal_store(val, out + (size_t)row * M + col);
            }
        }
    }
}

// ---------------------------------------------------------------------------
extern "C" void kernel_launch(void* const* d_in, const int* in_sizes, int n_in,
                              void* d_out, int out_size, void* d_ws, size_t ws_size,
                              hipStream_t stream) {
    const float* x      = (const float*)d_in[0];   // [N, 64]
    const float* y      = (const float*)d_in[1];   // [M, 64]
    const float* log_l  = (const float*)d_in[2];   // [64]
    const float* log_os = (const float*)d_in[3];   // [1]
    float* out = (float*)d_out;                    // [N, M]

    const int N = in_sizes[0] / D_DIM;             // 8192
    const int M = in_sizes[1] / D_DIM;             // 8192

    // workspace: xs_h | xs_l | ys_h | ys_l (f16) then x2 | y2 (f32), ~4.3 MB
    _Float16* xs_h = (_Float16*)d_ws;
    _Float16* xs_l = xs_h + (size_t)N * D_DIM;
    _Float16* ys_h = xs_l + (size_t)N * D_DIM;
    _Float16* ys_l = ys_h + (size_t)M * D_DIM;
    float*    x2   = (float*)(ys_l + (size_t)M * D_DIM);
    float*    y2   = x2 + N;

    ard_prep_kernel<<<N + M, 64, 0, stream>>>(x, y, log_l, N,
                                              xs_h, xs_l, ys_h, ys_l, x2, y2);

    dim3 grid(M / TILE, N / TILE);                 // 64 x 64 workgroups
    ard_rbf_wmma_kernel<<<grid, 256, 0, stream>>>(xs_h, xs_l, ys_h, ys_l,
                                                  x2, y2, log_os, out, M);
}
